// EdgeAwareBiLSTM_87462714016581
// MI455X (gfx1250) — matrix operand order repaired and checked by reading, not compiled, not run on hardware
//
#include <hip/hip_runtime.h>
#include <hip/hip_bf16.h>

typedef __attribute__((ext_vector_type(16))) _Float16 v16h;
typedef __attribute__((ext_vector_type(8)))  float    v8f;

#define B_  8
#define L_  4096
#define D_  256
#define H_  128
#define G_  512      // 4*H
#define BL_ 32768    // B*L

// ---- WMMA helpers (CDNA5: 16x16x32 f16 -> f32) -----------------------------

static __device__ inline v8f wmma_f16(v16h a, v16h b, v8f c) {
  return __builtin_amdgcn_wmma_f32_16x16x32_f16(
      /*neg_a=*/false, a, /*neg_b=*/false, b,
      /*c_mod=*/(short)0, c, /*reuse_a=*/false, /*reuse_b=*/false);
}

// A operand (16x32 f16 tile). Per lane (half = lane>=16):
//   elements 0..15 <- K = base+0..15 (half==0) / base+16..31 (half==1)
// caller passes p already offset by (half ? 16 : 0).
static __device__ inline v16h ldA_f16(const _Float16* __restrict__ p) {
  v16h a;
#pragma unroll
  for (int i = 0; i < 16; ++i) a[i] = p[i];
  return a;
}
// B operand (32x16 f16 tile). Per lane: column N = lane%16,
// elements 0..15 <- 16 consecutive K (caller offsets by (half ? 16 : 0)).
static __device__ inline v16h ldB_f16(const _Float16* __restrict__ p) {
  v16h b;
#pragma unroll
  for (int i = 0; i < 16; ++i) b[i] = p[i];
  return b;
}
// Same but converting from fp32 (used once for the scan's resident Whh).
static __device__ inline v16h ldB_f32(const float* __restrict__ p) {
  v16h b;
#pragma unroll
  for (int i = 0; i < 16; ++i) b[i] = (_Float16)p[i];
  return b;
}

static __device__ inline float sigm(float x) { return 1.f / (1.f + __expf(-x)); }

// ---- Kernel 0: bulk f32 -> f16 conversion (grid-stride over float4) --------
__global__ __launch_bounds__(256)
void cvt_f16_kernel(const float* __restrict__ src, _Float16* __restrict__ dst, int n4) {
  for (int i = blockIdx.x * blockDim.x + threadIdx.x; i < n4;
       i += gridDim.x * blockDim.x) {
    const float4 v = ((const float4*)src)[i];
    dst[4 * i + 0] = (_Float16)v.x;
    dst[4 * i + 1] = (_Float16)v.y;
    dst[4 * i + 2] = (_Float16)v.z;
    dst[4 * i + 3] = (_Float16)v.w;
  }
}

// ---- Kernel 1: x = features @ Wp.T + bp  -> f16 [BL][H] --------------------
// Each wave computes a 16x64 tile (4 N-subtiles): A chunk reused 4x.
__global__ __launch_bounds__(256)
void proj_kernel(const _Float16* __restrict__ F16, const _Float16* __restrict__ Wp16,
                 const float* __restrict__ bp, _Float16* __restrict__ xh) {
  const int lane  = threadIdx.x & 31;
  const int gw    = blockIdx.x * (blockDim.x >> 5) + (threadIdx.x >> 5);
  const int ntg   = gw & 1;          // 2 groups of 4 N-tiles (H/16 = 8)
  const int mtile = gw >> 1;         // BL/16 = 2048
  const int half  = lane >> 4;
  const int lm    = lane & 15;
  const int mrow  = mtile * 16 + lm;
  const int ncol0 = ntg * 64 + lm;   // + g*16
  v8f acc[4] = {};
#pragma unroll
  for (int kb = 0; kb < D_; kb += 32) {
    v16h a = ldA_f16(F16 + (size_t)mrow * D_ + kb + (half ? 16 : 0));
#pragma unroll
    for (int g = 0; g < 4; ++g) {
      v16h b = ldB_f16(Wp16 + (size_t)(ncol0 + g * 16) * D_ + kb + (half ? 16 : 0));
      acc[g] = wmma_f16(a, b, acc[g]);
    }
  }
#pragma unroll
  for (int g = 0; g < 4; ++g) {
    const int ncol = ncol0 + g * 16;
    const float bias = bp[ncol];
#pragma unroll
    for (int r = 0; r < 8; ++r) {
      const int m = mtile * 16 + r + (half ? 8 : 0);
      xh[(size_t)m * H_ + ncol] = (_Float16)(acc[g][r] + bias);
    }
  }
}

// ---- Kernel 2: gates = x @ Wih.T + (bih+bhh) -> f32, TIME-MAJOR [L][B][G] --
__global__ __launch_bounds__(256)
void gates_kernel(const _Float16* __restrict__ xh,
                  const _Float16* __restrict__ Wih16_f, const float* __restrict__ bih_f,
                  const float* __restrict__ bhh_f,
                  const _Float16* __restrict__ Wih16_b, const float* __restrict__ bih_b,
                  const float* __restrict__ bhh_b,
                  float* __restrict__ gates_f, float* __restrict__ gates_b) {
  const int dir = blockIdx.y;
  const _Float16* Wih = dir ? Wih16_b : Wih16_f;
  const float* bih = dir ? bih_b : bih_f;
  const float* bhh = dir ? bhh_b : bhh_f;
  float* gates = dir ? gates_b : gates_f;

  const int lane  = threadIdx.x & 31;
  const int gw    = blockIdx.x * (blockDim.x >> 5) + (threadIdx.x >> 5);
  const int ntg   = gw & 7;          // 8 groups of 4 N-tiles (G/16 = 32)
  const int mtile = gw >> 3;         // BL/16 = 2048
  const int half  = lane >> 4;
  const int lm    = lane & 15;
  const int mrow  = mtile * 16 + lm;
  const int ncol0 = ntg * 64 + lm;

  v8f acc[4];
#pragma unroll
  for (int g = 0; g < 4; ++g) {
    const float bias = bih[ncol0 + g * 16] + bhh[ncol0 + g * 16];
#pragma unroll
    for (int r = 0; r < 8; ++r) acc[g][r] = bias;
  }
#pragma unroll
  for (int kb = 0; kb < H_; kb += 32) {
    v16h a = ldA_f16(xh + (size_t)mrow * H_ + kb + (half ? 16 : 0));
#pragma unroll
    for (int g = 0; g < 4; ++g) {
      v16h b = ldB_f16(Wih + (size_t)(ncol0 + g * 16) * H_ + kb + (half ? 16 : 0));
      acc[g] = wmma_f16(a, b, acc[g]);
    }
  }
#pragma unroll
  for (int g = 0; g < 4; ++g) {
#pragma unroll
    for (int r = 0; r < 8; ++r) {
      const int m = mtile * 16 + r + (half ? 8 : 0);
      const int b = m >> 12;           // batch  (L = 4096 = 2^12)
      const int l = m & (L_ - 1);      // timestep
      gates[((size_t)l * B_ + b) * G_ + ncol0 + g * 16] = acc[g][r];
    }
  }
}

// ---- Kernel 3: sequential LSTM scan, one workgroup per direction -----------
__global__ __launch_bounds__(1024, 1)
void lstm_kernel(const float* __restrict__ Whh_f, const float* __restrict__ Whh_b,
                 const float* __restrict__ gates_f, const float* __restrict__ gates_b,
                 _Float16* __restrict__ hf, _Float16* __restrict__ hb) {
  __shared__ _Float16 hbuf[16 * H_];   // h in WMMA-A row layout (rows 8..15 = 0)
  __shared__ float    gbuf[8 * G_];    // gate preactivations for this step

  const int dir = blockIdx.x;
  const float* Whh   = dir ? Whh_b   : Whh_f;
  const float* gates = dir ? gates_b : gates_f;
  _Float16*    hout  = dir ? hb      : hf;

  const int tid  = threadIdx.x;
  const int lane = tid & 31;
  const int wave = tid >> 5;           // N-tile id, 0..31
  const int half = lane >> 4;
  const int lm   = lane & 15;
  const int gcol = wave * 16 + lm;
  const int half8  = half ? 16 : 0;

  // Persistent B operands: Whh[gcol][0..127] as 4 x v16h (converted once)
  v16h wB[4];
#pragma unroll
  for (int k = 0; k < 4; ++k)
    wB[k] = ldB_f32(Whh + (size_t)gcol * H_ + k * 32 + (half ? 16 : 0));

  hbuf[tid]        = (_Float16)0.f;
  hbuf[tid + 1024] = (_Float16)0.f;

  const int eb = tid >> 7;             // batch 0..7
  const int ej = tid & 127;            // hidden unit 0..127
  float c = 0.f;                       // cell state, register-resident

  // software pipeline: C-init gate values for the upcoming step
  float gc[8];
  {
    const int t0 = dir ? (L_ - 1) : 0;
    if (lane < 16) {
#pragma unroll
      for (int r = 0; r < 8; ++r)
        gc[r] = gates[((size_t)t0 * B_ + r) * G_ + gcol];
    }
  }
  __syncthreads();

  for (int s = 0; s < L_; ++s) {
    const int t = dir ? (L_ - 1 - s) : s;

    // prefetch the gate block ~8 steps ahead into cache (global_prefetch_b8)
    const int sp = (s + 8 < L_) ? s + 8 : L_ - 1;
    const int tp = dir ? (L_ - 1 - sp) : sp;
    if (tid < 128)
      __builtin_prefetch(gates + (size_t)tp * B_ * G_ + tid * 32, 0, 1);

    // C init from the pipelined registers
    v8f acc0 = {};
    if (lane < 16) {
#pragma unroll
      for (int r = 0; r < 8; ++r) acc0[r] = gc[r];
    }
    // kick off next step's gate loads (hidden behind WMMA + barriers)
    const int sn = (s + 1 < L_) ? s + 1 : s;
    const int tn = dir ? (L_ - 1 - sn) : sn;
    if (lane < 16) {
#pragma unroll
      for (int r = 0; r < 8; ++r)
        gc[r] = gates[((size_t)tn * B_ + r) * G_ + gcol];
    }

    // A operands: previous h from LDS
    v16h a0 = ldA_f16(hbuf + lm * H_ +  0 + half8);
    v16h a1 = ldA_f16(hbuf + lm * H_ + 32 + half8);
    v16h a2 = ldA_f16(hbuf + lm * H_ + 64 + half8);
    v16h a3 = ldA_f16(hbuf + lm * H_ + 96 + half8);

    // two independent WMMA chains to halve the RAW-hazard depth
    v8f acc1 = {};
    acc0 = wmma_f16(a0, wB[0], acc0);
    acc1 = wmma_f16(a1, wB[1], acc1);
    acc0 = wmma_f16(a2, wB[2], acc0);
    acc1 = wmma_f16(a3, wB[3], acc1);

    if (lane < 16) {
#pragma unroll
      for (int r = 0; r < 8; ++r) gbuf[r * G_ + gcol] = acc0[r] + acc1[r];
    }
    __syncthreads();

    // pointwise cell update: gate order i, f, g, o
    const float gi = gbuf[eb * G_ + ej];
    const float gf = gbuf[eb * G_ + H_ + ej];
    const float gg = gbuf[eb * G_ + 2 * H_ + ej];
    const float go = gbuf[eb * G_ + 3 * H_ + ej];
    c = sigm(gf) * c + sigm(gi) * tanhf(gg);
    const float h = sigm(go) * tanhf(c);
    hbuf[eb * H_ + ej] = (_Float16)h;
    hout[((size_t)eb * L_ + t) * H_ + ej] = (_Float16)h;
    __syncthreads();
  }
}

// ---- Kernel 4: hid = relu([hf|hb] @ W1.T + b1) -> f16 [BL][H] --------------
__global__ __launch_bounds__(256)
void head1_kernel(const _Float16* __restrict__ hf, const _Float16* __restrict__ hb,
                  const _Float16* __restrict__ W116, const float* __restrict__ b1,
                  _Float16* __restrict__ hid) {
  const int lane  = threadIdx.x & 31;
  const int gw    = blockIdx.x * (blockDim.x >> 5) + (threadIdx.x >> 5);
  const int ntg   = gw & 1;          // 2 groups of 4 N-tiles (H/16 = 8)
  const int mtile = gw >> 1;         // BL/16
  const int half  = lane >> 4;
  const int lm    = lane & 15;
  const int mrow  = mtile * 16 + lm;
  const int ncol0 = ntg * 64 + lm;
  v8f acc[4];
#pragma unroll
  for (int g = 0; g < 4; ++g) {
    const float bias = b1[ncol0 + g * 16];
#pragma unroll
    for (int r = 0; r < 8; ++r) acc[g][r] = bias;
  }
#pragma unroll
  for (int kb = 0; kb < 2 * H_; kb += 32) {
    const _Float16* src = (kb < H_) ? (hf + (size_t)mrow * H_ + kb)
                                    : (hb + (size_t)mrow * H_ + (kb - H_));
    v16h a = ldA_f16(src + (half ? 16 : 0));
#pragma unroll
    for (int g = 0; g < 4; ++g) {
      v16h b = ldB_f16(W116 + (size_t)(ncol0 + g * 16) * (2 * H_) + kb + (half ? 16 : 0));
      acc[g] = wmma_f16(a, b, acc[g]);
    }
  }
#pragma unroll
  for (int g = 0; g < 4; ++g) {
#pragma unroll
    for (int r = 0; r < 8; ++r) {
      const int m = mtile * 16 + r + (half ? 8 : 0);
      hid[(size_t)m * H_ + ncol0 + g * 16] = (_Float16)fmaxf(acc[g][r], 0.f);
    }
  }
}

// ---- Kernel 5: probs = clip(sigmoid(hid @ W2.T + b2) * edge_boost, 0, 1) ---
__global__ __launch_bounds__(256)
void head2_kernel(const _Float16* __restrict__ hid, const float* __restrict__ W2,
                  const float* __restrict__ b2, float* __restrict__ probs) {
  const int m = blockIdx.x * blockDim.x + threadIdx.x;   // m = b*L + l
  float dot = b2[0];
#pragma unroll 8
  for (int j = 0; j < H_; ++j)
    dot += (float)hid[(size_t)m * H_ + j] * W2[j];
  float p = sigm(dot);
  const int l = m & (L_ - 1);
  if (l == 0 || l == L_ - 1) p *= 1.2f;
  probs[m] = fminf(fmaxf(p, 0.f), 1.f);
}

// ---- Kernel 6: adjacent-gradient comparison --------------------------------
__global__ __launch_bounds__(256)
void adj_kernel(const float* __restrict__ probs, float* __restrict__ adj) {
  const int m = blockIdx.x * blockDim.x + threadIdx.x;
  const int l = m & (L_ - 1);
  float v = 0.f;
  if (l > 0 && l < L_ - 1) {
    const float pm = probs[m - 1], pc = probs[m], pp = probs[m + 1];
    const float lg = pc - pm, rg = pc - pp;
    const float alg = fabsf(lg), arg = fabsf(rg);
    if (lg < 0.f && alg > arg)      v = -1.f;
    else if (rg < 0.f && arg > alg) v = 1.f;
  }
  adj[m] = v;
}

// ---- host-side launcher ----------------------------------------------------
extern "C" void kernel_launch(void* const* d_in, const int* in_sizes, int n_in,
                              void* d_out, int out_size, void* d_ws, size_t ws_size,
                              hipStream_t stream) {
  (void)in_sizes; (void)n_in; (void)out_size; (void)ws_size;
  const float* F     = (const float*)d_in[0];
  const float* Wp    = (const float*)d_in[1];
  const float* bp    = (const float*)d_in[2];
  const float* Wih_f = (const float*)d_in[3];
  const float* Whh_f = (const float*)d_in[4];
  const float* bih_f = (const float*)d_in[5];
  const float* bhh_f = (const float*)d_in[6];
  const float* Wih_b = (const float*)d_in[7];
  const float* Whh_b = (const float*)d_in[8];
  const float* bih_b = (const float*)d_in[9];
  const float* bhh_b = (const float*)d_in[10];
  const float* W1    = (const float*)d_in[11];
  const float* b1    = (const float*)d_in[12];
  const float* W2    = (const float*)d_in[13];
  const float* b2    = (const float*)d_in[14];

  char* ws = (char*)d_ws;
  size_t off = 0;
  _Float16* xh      = (_Float16*)(ws + off); off += (size_t)BL_ * H_ * 2;      //   8 MB
  float*    gates_f = (float*)   (ws + off); off += (size_t)BL_ * G_ * 4;      //  64 MB
  float*    gates_b = (float*)   (ws + off); off += (size_t)BL_ * G_ * 4;      //  64 MB
  _Float16* hf      = (_Float16*)(ws + off); off += (size_t)BL_ * H_ * 2;      //   8 MB
  _Float16* hb      = (_Float16*)(ws + off); off += (size_t)BL_ * H_ * 2;      //   8 MB
  _Float16* hid     = (_Float16*)(ws + off); off += (size_t)BL_ * H_ * 2;      //   8 MB
  _Float16* F16     = (_Float16*)(ws + off); off += (size_t)BL_ * D_ * 2;      //  16 MB
  _Float16* Wp16    = (_Float16*)(ws + off); off += (size_t)H_ * D_ * 2;
  _Float16* Wih16_f = (_Float16*)(ws + off); off += (size_t)G_ * H_ * 2;
  _Float16* Wih16_b = (_Float16*)(ws + off); off += (size_t)G_ * H_ * 2;
  _Float16* W116    = (_Float16*)(ws + off); off += (size_t)H_ * 2 * H_ * 2;

  float* probs = (float*)d_out;       // [B, L]
  float* adj   = probs + BL_;         // [B, L]

  // one-time f16 conversions
  cvt_f16_kernel<<<8192, 256, 0, stream>>>(F,     F16,     (BL_ * D_) / 4);
  cvt_f16_kernel<<<32,   256, 0, stream>>>(Wp,    Wp16,    (H_ * D_) / 4);
  cvt_f16_kernel<<<64,   256, 0, stream>>>(Wih_f, Wih16_f, (G_ * H_) / 4);
  cvt_f16_kernel<<<64,   256, 0, stream>>>(Wih_b, Wih16_b, (G_ * H_) / 4);
  cvt_f16_kernel<<<32,   256, 0, stream>>>(W1,    W116,    (H_ * 2 * H_) / 4);

  proj_kernel <<<512, 256, 0, stream>>>(F16, Wp16, bp, xh);                 // 2048x2 waves
  gates_kernel<<<dim3(2048, 2), 256, 0, stream>>>(xh, Wih16_f, bih_f, bhh_f,
                                                  Wih16_b, bih_b, bhh_b,
                                                  gates_f, gates_b);        // 2048x8 waves/dir
  lstm_kernel <<<2, 1024, 0, stream>>>(Whh_f, Whh_b, gates_f, gates_b, hf, hb);
  head1_kernel<<<512, 256, 0, stream>>>(hf, hb, W116, b1, hid);
  head2_kernel<<<BL_ / 256, 256, 0, stream>>>(hid, W2, b2, probs);
  adj_kernel  <<<BL_ / 256, 256, 0, stream>>>(probs, adj);
}
